// MultiHeadAttention_27470610825944
// MI455X (gfx1250) — matrix operand order repaired and checked by reading, not compiled, not run on hardware
//
#include <hip/hip_runtime.h>
#include <hip/hip_bf16.h>
#include <math.h>
#include <stdint.h>

typedef __attribute__((ext_vector_type(16))) _Float16 v16h;
typedef __attribute__((ext_vector_type(8)))  _Float16 v8h;
typedef __attribute__((ext_vector_type(4)))  _Float16 v4h;
typedef __attribute__((ext_vector_type(8)))  float    v8f;

#define B_  4
#define S_  2048
#define D_  768
#define H_  12
#define DK_ 64

// LDS row pitches (halves): padded so the 16-lane bank pattern is a full
// permutation (stride*16 != 0 mod 64 banks), while keeping 16B alignment.
#define PK_  72   // 128x64 key tile / 128x64 P tile / 64x64 V tile pitch
#define PW_  40   // 128x32 weight tile pitch

// ---------------------------------------------------------------------------
// gfx1250 async global->LDS copy (ASYNCcnt path), with safe fallback.
// Probe-verified: builtin takes (global v4i*, local v4i*, imm offset, cpol).
// ---------------------------------------------------------------------------
#if defined(__AMDGCN__) && __has_builtin(__builtin_amdgcn_global_load_async_to_lds_b128)
#define HAVE_ASYNC_LDS 1
#else
#define HAVE_ASYNC_LDS 0
#endif

#if HAVE_ASYNC_LDS
typedef int v4i_ __attribute__((vector_size(16)));
typedef __attribute__((address_space(1))) v4i_ GV4;
typedef __attribute__((address_space(3))) v4i_ LV4;
#endif

static __device__ __forceinline__ void copy16_g2lds(void* lds_dst, const void* gsrc) {
#if HAVE_ASYNC_LDS
  __builtin_amdgcn_global_load_async_to_lds_b128(
      (GV4*)(uintptr_t)gsrc,
      (LV4*)(uint32_t)(uintptr_t)lds_dst,
      0, 0);
#else
  *(uint4*)lds_dst = *(const uint4*)gsrc;
#endif
}

static __device__ __forceinline__ void wait_async() {
#if HAVE_ASYNC_LDS
#if __has_builtin(__builtin_amdgcn_s_wait_asynccnt)
  __builtin_amdgcn_s_wait_asynccnt(0);
#else
  asm volatile("s_wait_asynccnt 0" ::: "memory");
#endif
#endif
}

// ---------------------------------------------------------------------------
// WMMA wrapper: D = A(16x32 f16) * B(32x16 f16) + C(16x16 f32)
// ---------------------------------------------------------------------------
static __device__ __forceinline__ v8f wmma16(v16h a, v16h b, v8f c) {
  return __builtin_amdgcn_wmma_f32_16x16x32_f16(
      false, a, false, b, (short)0, c, false, false);
}

// ---------------------------------------------------------------------------
// Fragment loaders per CDNA5 ISA §7.12.2 (wave32). Per-lane data is two
// contiguous 8-half runs -> two 16B vector loads each (b128 path).
// ---------------------------------------------------------------------------
static __device__ __forceinline__ v16h load_a_f16(const _Float16* __restrict__ src,
                                                  int ld, int m0, int k0, int lane) {
  const int m  = m0 + (lane & 15);
  const int kb = k0 + ((lane >> 4) << 3);        // +8 for upper half-wave
  const _Float16* p = src + (size_t)m * ld + kb;
  const v8h lo = *(const v8h*)p;                 // K = kb .. kb+7
  const v8h hi = *(const v8h*)(p + 16);          // K = kb+16 .. kb+23
  return __builtin_shufflevector(lo, hi, 0, 1, 2, 3, 4, 5, 6, 7,
                                         8, 9, 10, 11, 12, 13, 14, 15);
}

// B 32x16 from [N][K]-major source: lane 0-15 -> N=lane, K=k0..k0+7,k0+16..k0+23;
// lane 16-31 -> N=lane-16, K=k0+8..k0+15,k0+24..k0+31. Matches A fragment order.
static __device__ __forceinline__ v16h load_b_nk_f16(const _Float16* __restrict__ src,
                                                     int ld, int k0, int n0, int lane) {
  const int n  = n0 + (lane & 15);
  const int kb = k0 + ((lane >> 4) << 3);
  const _Float16* p = src + (size_t)n * ld + kb;
  const v8h lo = *(const v8h*)p;
  const v8h hi = *(const v8h*)(p + 16);
  return __builtin_shufflevector(lo, hi, 0, 1, 2, 3, 4, 5, 6, 7,
                                         8, 9, 10, 11, 12, 13, 14, 15);
}

// ---------------------------------------------------------------------------
// Kernel 0a: fp32 -> f16 elementwise conversion (x4 vectorized)
// ---------------------------------------------------------------------------
__global__ __launch_bounds__(256) void cvt16_kernel(const float* __restrict__ src,
                                                    _Float16* __restrict__ dst, int n4) {
  const int i = blockIdx.x * 256 + threadIdx.x;
  if (i < n4) {
    const float4 f = ((const float4*)src)[i];
    v4h h;
    h[0] = (_Float16)f.x; h[1] = (_Float16)f.y;
    h[2] = (_Float16)f.z; h[3] = (_Float16)f.w;
    ((v4h*)dst)[i] = h;
  }
}

// ---------------------------------------------------------------------------
// Kernel 0b: fp32 [K][N] -> f16 [N][K] transpose+convert (32x32 LDS tiles).
// grid=(24,24) block=256 (32x8).
// ---------------------------------------------------------------------------
__global__ __launch_bounds__(256) void cvtT_kernel(const float* __restrict__ src,
                                                   _Float16* __restrict__ dst) {
  __shared__ _Float16 tile[32][33];
  const int lx = threadIdx.x & 31, ly = threadIdx.x >> 5;
#pragma unroll
  for (int i = 0; i < 4; ++i) {
    const int k = blockIdx.y * 32 + ly + i * 8;
    const int n = blockIdx.x * 32 + lx;
    tile[ly + i * 8][lx] = (_Float16)src[(size_t)k * D_ + n];
  }
  __syncthreads();
#pragma unroll
  for (int i = 0; i < 4; ++i) {
    const int n = blockIdx.x * 32 + ly + i * 8;
    const int k = blockIdx.y * 32 + lx;
    dst[(size_t)n * D_ + k] = tile[lx][ly + i * 8];
  }
}

// ---------------------------------------------------------------------------
// Kernel 1: fused QKV projection (Wt = f16 [N][K], tile via async->LDS).
// grid=(6,64,3) block=256.  q,k -> [B,H,S,64]; v -> TRANSPOSED [B,H,64,S].
// ---------------------------------------------------------------------------
__global__ __launch_bounds__(256) void qkv_proj_kernel(
    const _Float16* __restrict__ xq, const _Float16* __restrict__ xk,
    const _Float16* __restrict__ xv,
    const _Float16* __restrict__ wqt, const float* __restrict__ bq,
    const _Float16* __restrict__ wkt, const float* __restrict__ bk,
    const _Float16* __restrict__ wvt, const float* __restrict__ bv,
    _Float16* __restrict__ qbuf, _Float16* __restrict__ kbuf,
    _Float16* __restrict__ vbuf) {
  __shared__ _Float16 sm_bt[128 * PW_];               // [n][k] tile, 10 KB
  const int tid = threadIdx.x, lane = tid & 31, w = tid >> 5;
  const int wr = w >> 1, wc = w & 1;
  const _Float16* X; const _Float16* Wt; const float* bias; _Float16* out; float scale;
  if (blockIdx.z == 0)      { X = xq; Wt = wqt; bias = bq; out = qbuf; scale = 0.125f; }
  else if (blockIdx.z == 1) { X = xk; Wt = wkt; bias = bk; out = kbuf; scale = 1.0f; }
  else                      { X = xv; Wt = wvt; bias = bv; out = vbuf; scale = 1.0f; }
  const bool vtrans = (blockIdx.z == 2);
  const int m0   = blockIdx.y * 128 + wr * 32;
  const int nblk = blockIdx.x * 128;
  const int n0   = nblk + wc * 64;

  const v8f z8 = {0.f, 0.f, 0.f, 0.f, 0.f, 0.f, 0.f, 0.f};
  v8f c[2][4];
#pragma unroll
  for (int i = 0; i < 2; ++i)
#pragma unroll
    for (int j = 0; j < 4; ++j) c[i][j] = z8;

  for (int kt = 0; kt < D_ / 32; ++kt) {
    const int k0 = kt * 32;
    // stage Wt tile [128 n][32 k]: 512 x 16B segments, straight-line
    {
      const int s0 = tid, s1 = tid + 256;
      copy16_g2lds(sm_bt + (s0 >> 2) * PW_ + (s0 & 3) * 8,
                   Wt + (size_t)(nblk + (s0 >> 2)) * D_ + k0 + (s0 & 3) * 8);
      copy16_g2lds(sm_bt + (s1 >> 2) * PW_ + (s1 & 3) * 8,
                   Wt + (size_t)(nblk + (s1 >> 2)) * D_ + k0 + (s1 & 3) * 8);
    }
    v16h a0 = load_a_f16(X, D_, m0,      k0, lane);
    v16h a1 = load_a_f16(X, D_, m0 + 16, k0, lane);
    wait_async();
    __syncthreads();
#pragma unroll
    for (int nt = 0; nt < 4; ++nt) {
      v16h bf = load_b_nk_f16(sm_bt, PW_, 0, wc * 64 + nt * 16, lane);
      c[0][nt] = wmma16(a0, bf, c[0][nt]);
      c[1][nt] = wmma16(a1, bf, c[1][nt]);
    }
    __syncthreads();
  }

  const int ml = (lane >> 4) << 3, nl = lane & 15;
#pragma unroll
  for (int mt = 0; mt < 2; ++mt)
#pragma unroll
    for (int nt = 0; nt < 4; ++nt)
#pragma unroll
      for (int r = 0; r < 8; ++r) {
        const int row = m0 + mt * 16 + r + ml;
        const int col = n0 + nt * 16 + nl;
        const float val = (c[mt][nt][r] + bias[col]) * scale;
        const int b = row >> 11, s = row & (S_ - 1);
        const int h = col >> 6,  d = col & 63;
        const size_t oi = vtrans
            ? ((size_t)(b * H_ + h) * DK_ + d) * S_ + s    // v: [B,H,dv,S]
            : ((size_t)(b * H_ + h) * S_ + s) * DK_ + d;   // q,k: [B,H,S,dk]
        out[oi] = (_Float16)val;
      }
}

// ---------------------------------------------------------------------------
// Kernel 2a: scores = q @ k^T (per b,h), masked -> attn region (raw).
// grid=(16,16,48) block=256.  key tile via async->LDS.
// ---------------------------------------------------------------------------
__global__ __launch_bounds__(256) void scores_kernel(
    const _Float16* __restrict__ qbuf, const _Float16* __restrict__ kbuf,
    const unsigned char* __restrict__ mask, float* __restrict__ attn) {
  __shared__ _Float16 sm_k[128 * PK_];                // 18 KB key tile
  const int tid = threadIdx.x, lane = tid & 31, w = tid >> 5;
  const int wr = w >> 1, wc = w & 1;
  const int bh = blockIdx.z, b = bh / H_;
  const int m0   = blockIdx.y * 128 + wr * 32;        // query rows
  const int nblk = blockIdx.x * 128;                  // key block base
  const _Float16* qp = qbuf + (size_t)bh * S_ * DK_;
  const _Float16* kp = kbuf + (size_t)bh * S_ * DK_;

  // stage 128 keys x 64 dk: 1024 x 16B segments, straight-line
#pragma unroll
  for (int j = 0; j < 4; ++j) {
    const int s = tid + j * 256;
    copy16_g2lds(sm_k + (s >> 3) * PK_ + (s & 7) * 8,
                 kp + (size_t)(nblk + (s >> 3)) * DK_ + (s & 7) * 8);
  }

  const v8f z8 = {0.f, 0.f, 0.f, 0.f, 0.f, 0.f, 0.f, 0.f};
  v8f c[2][4];
#pragma unroll
  for (int i = 0; i < 2; ++i)
#pragma unroll
    for (int j = 0; j < 4; ++j) c[i][j] = z8;

  wait_async();
  __syncthreads();

#pragma unroll
  for (int kt = 0; kt < 2; ++kt) {
    const int k0 = kt * 32;
    v16h a0 = load_a_f16(qp, DK_, m0,      k0, lane);
    v16h a1 = load_a_f16(qp, DK_, m0 + 16, k0, lane);
#pragma unroll
    for (int nt = 0; nt < 4; ++nt) {
      v16h bf = load_b_nk_f16(sm_k, PK_, k0, wc * 64 + nt * 16, lane);
      c[0][nt] = wmma16(a0, bf, c[0][nt]);
      c[1][nt] = wmma16(a1, bf, c[1][nt]);
    }
  }

  float* ap = attn + (size_t)bh * S_ * S_;
  const unsigned char* mp = mask + (size_t)b * S_ * S_;
  const int ml = (lane >> 4) << 3, nl = lane & 15;
#pragma unroll
  for (int mt = 0; mt < 2; ++mt)
#pragma unroll
    for (int nt = 0; nt < 4; ++nt)
#pragma unroll
      for (int r = 0; r < 8; ++r) {
        const int sq = m0 + mt * 16 + r + ml;
        const int sk = nblk + wc * 64 + nt * 16 + nl;
        float v = c[mt][nt][r];
        if (mp[(size_t)sq * S_ + sk]) v = -1e9f;
        ap[(size_t)sq * S_ + sk] = v;
      }
}

// ---------------------------------------------------------------------------
// Kernel 2b: softmax over full rows + context = P @ v. grid=(16,48) block=256
// vt is TRANSPOSED [B,H,dv,S]. Async V-chunk copy overlaps the exp work.
// ---------------------------------------------------------------------------
__global__ __launch_bounds__(256) void softmax_ctx_kernel(
    const _Float16* __restrict__ vt, float* __restrict__ attn,
    _Float16* __restrict__ ctx) {
  __shared__ float    sm_max[128];
  __shared__ float    sm_inv[128];
  __shared__ _Float16 sm_p[128 * PK_];                // 18 KB P tile [q][key]
  __shared__ _Float16 sm_v[64 * PK_];                 //  9 KB V tile [dv][key]

  const int tid = threadIdx.x, lane = tid & 31, w = tid >> 5;
  const int bh = blockIdx.y, b = bh / H_, h = bh % H_;
  const int qbase = blockIdx.x * 128;
  float* ap = attn + (size_t)bh * S_ * S_;

  // --- phase 1: row max & sum-exp (wave-cooperative, shfl reductions) ---
  for (int rr = 0; rr < 16; ++rr) {
    const int rloc = w * 16 + rr;
    const float* sp = ap + (size_t)(qbase + rloc) * S_;
    float mx = -3.0e38f;
    for (int i = 0; i < S_ / 32; ++i) mx = fmaxf(mx, sp[i * 32 + lane]);
#pragma unroll
    for (int off = 16; off > 0; off >>= 1) mx = fmaxf(mx, __shfl_xor(mx, off, 32));
    float sum = 0.f;
    for (int i = 0; i < S_ / 32; ++i) sum += __expf(sp[i * 32 + lane] - mx);
#pragma unroll
    for (int off = 16; off > 0; off >>= 1) sum += __shfl_xor(sum, off, 32);
    if (lane == 0) { sm_max[rloc] = mx; sm_inv[rloc] = 1.0f / sum; }
  }
  __syncthreads();

  const _Float16* vp = vt + (size_t)bh * DK_ * S_;    // [dv][S]
  const v8f z8 = {0.f, 0.f, 0.f, 0.f, 0.f, 0.f, 0.f, 0.f};
  v8f acc[4];
#pragma unroll
  for (int j = 0; j < 4; ++j) acc[j] = z8;

  for (int ch = 0; ch < S_ / 64; ++ch) {
    // async copy of V chunk [64 dv][64 keys] -> overlaps exp work below
    {
      const int s0 = tid, s1 = tid + 256;             // 512 x 16B segments
      copy16_g2lds(sm_v + (s0 >> 3) * PK_ + (s0 & 7) * 8,
                   vp + (size_t)(s0 >> 3) * S_ + ch * 64 + (s0 & 7) * 8);
      copy16_g2lds(sm_v + (s1 >> 3) * PK_ + (s1 & 7) * 8,
                   vp + (size_t)(s1 >> 3) * S_ + ch * 64 + (s1 & 7) * 8);
    }
    // normalize chunk: write final P (fp32) + stage f16 P tile
#pragma unroll 4
    for (int i = 0; i < 32; ++i) {                    // 256 thr * 32 = 128x64
      const int lin = i * 256 + tid;
      const int rl = lin >> 6, cl = lin & 63;
      const size_t gi = (size_t)(qbase + rl) * S_ + ch * 64 + cl;
      const float p = __expf(ap[gi] - sm_max[rl]) * sm_inv[rl];
      ap[gi] = p;
      sm_p[rl * PK_ + cl] = (_Float16)p;
    }
    wait_async();
    __syncthreads();
#pragma unroll
    for (int ks = 0; ks < 2; ++ks) {
      v16h a = load_a_f16(sm_p, PK_, w * 16, ks * 32, lane);
#pragma unroll
      for (int nt = 0; nt < 4; ++nt) {
        v16h bf = load_b_nk_f16(sm_v, PK_, ks * 32, nt * 16, lane);
        acc[nt] = wmma16(a, bf, acc[nt]);
      }
    }
    __syncthreads();
  }

  const int ml = (lane >> 4) << 3, nl = lane & 15;
#pragma unroll
  for (int nt = 0; nt < 4; ++nt)
#pragma unroll
    for (int r = 0; r < 8; ++r) {
      const int s = qbase + w * 16 + r + ml;
      const int d = nt * 16 + nl;
      ctx[((size_t)(b * S_ + s)) * D_ + h * DK_ + d] = (_Float16)acc[nt][r];
    }
}

// ---------------------------------------------------------------------------
// Kernel 3a: output projection + bias + residual -> fp32 scratch. grid=(6,64)
// ---------------------------------------------------------------------------
__global__ __launch_bounds__(256) void out_proj_kernel(
    const _Float16* __restrict__ ctx, const _Float16* __restrict__ wot,
    const float* __restrict__ bo, const float* __restrict__ Qin,
    float* __restrict__ xbuf) {
  __shared__ _Float16 sm_bt[128 * PW_];               // [n][k] tile, 10 KB
  const int tid = threadIdx.x, lane = tid & 31, w = tid >> 5;
  const int wr = w >> 1, wc = w & 1;
  const int m0   = blockIdx.y * 128 + wr * 32;
  const int nblk = blockIdx.x * 128;
  const int n0   = nblk + wc * 64;

  const v8f z8 = {0.f, 0.f, 0.f, 0.f, 0.f, 0.f, 0.f, 0.f};
  v8f c[2][4];
#pragma unroll
  for (int i = 0; i < 2; ++i)
#pragma unroll
    for (int j = 0; j < 4; ++j) c[i][j] = z8;

  for (int kt = 0; kt < D_ / 32; ++kt) {
    const int k0 = kt * 32;
    {
      const int s0 = tid, s1 = tid + 256;
      copy16_g2lds(sm_bt + (s0 >> 2) * PW_ + (s0 & 3) * 8,
                   wot + (size_t)(nblk + (s0 >> 2)) * D_ + k0 + (s0 & 3) * 8);
      copy16_g2lds(sm_bt + (s1 >> 2) * PW_ + (s1 & 3) * 8,
                   wot + (size_t)(nblk + (s1 >> 2)) * D_ + k0 + (s1 & 3) * 8);
    }
    v16h a0 = load_a_f16(ctx, D_, m0,      k0, lane);
    v16h a1 = load_a_f16(ctx, D_, m0 + 16, k0, lane);
    wait_async();
    __syncthreads();
#pragma unroll
    for (int nt = 0; nt < 4; ++nt) {
      v16h bf = load_b_nk_f16(sm_bt, PW_, 0, wc * 64 + nt * 16, lane);
      c[0][nt] = wmma16(a0, bf, c[0][nt]);
      c[1][nt] = wmma16(a1, bf, c[1][nt]);
    }
    __syncthreads();
  }

  const int ml = (lane >> 4) << 3, nl = lane & 15;
#pragma unroll
  for (int mt = 0; mt < 2; ++mt)
#pragma unroll
    for (int nt = 0; nt < 4; ++nt)
#pragma unroll
      for (int r = 0; r < 8; ++r) {
        const int row = m0 + mt * 16 + r + ml;
        const int col = n0 + nt * 16 + nl;
        const size_t gi = (size_t)row * D_ + col;
        xbuf[gi] = c[mt][nt][r] + bo[col] + Qin[gi];
      }
}

// ---------------------------------------------------------------------------
// Kernel 3b: LayerNorm over last dim. One wave per row. grid=1024 block=256
// ---------------------------------------------------------------------------
__global__ __launch_bounds__(256) void layernorm_kernel(
    const float* __restrict__ xbuf, const float* __restrict__ gamma,
    const float* __restrict__ beta, float* __restrict__ outp) {
  const int tid = threadIdx.x, lane = tid & 31, w = tid >> 5;
  const int row = blockIdx.x * 8 + w;
  const float* xp = xbuf + (size_t)row * D_;

  float s = 0.f;
  for (int i = 0; i < D_ / 32; ++i) s += xp[i * 32 + lane];
#pragma unroll
  for (int off = 16; off > 0; off >>= 1) s += __shfl_xor(s, off, 32);
  const float mean = s * (1.0f / D_);

  float vs = 0.f;
  for (int i = 0; i < D_ / 32; ++i) {
    const float d = xp[i * 32 + lane] - mean;
    vs += d * d;
  }
#pragma unroll
  for (int off = 16; off > 0; off >>= 1) vs += __shfl_xor(vs, off, 32);
  const float rstd = rsqrtf(vs * (1.0f / D_) + 1e-5f);

  float* op = outp + (size_t)row * D_;
  for (int i = 0; i < D_ / 32; ++i) {
    const int idx = i * 32 + lane;
    op[idx] = (xp[idx] - mean) * rstd * gamma[idx] + beta[idx];
  }
}

// ---------------------------------------------------------------------------
extern "C" void kernel_launch(void* const* d_in, const int* in_sizes, int n_in,
                              void* d_out, int out_size, void* d_ws, size_t ws_size,
                              hipStream_t stream) {
  (void)in_sizes; (void)n_in; (void)out_size; (void)ws_size;
  const float* Q  = (const float*)d_in[0];
  const float* K  = (const float*)d_in[1];
  const float* V  = (const float*)d_in[2];
  const unsigned char* mask = (const unsigned char*)d_in[3];
  const float* Wq = (const float*)d_in[4];
  const float* bq = (const float*)d_in[5];
  const float* Wk = (const float*)d_in[6];
  const float* bk = (const float*)d_in[7];
  const float* Wv = (const float*)d_in[8];
  const float* bv = (const float*)d_in[9];
  const float* Wo = (const float*)d_in[10];
  const float* bo = (const float*)d_in[11];
  const float* gamma = (const float*)d_in[12];
  const float* beta  = (const float*)d_in[13];

  float* outp = (float*)d_out;
  float* attn = outp + (size_t)B_ * S_ * D_;          // [B,H,S,S] fp32 region

  const size_t n_e = (size_t)B_ * S_ * D_;            // 6,291,456
  const size_t n_w = (size_t)D_ * D_;                 //   589,824
  _Float16* xq    = (_Float16*)d_ws;                  // f16 copies of Q,K,V
  _Float16* xk    = xq + n_e;
  _Float16* xv    = xk + n_e;
  _Float16* wq16t = xv + n_e;                         // f16 W^T weights
  _Float16* wk16t = wq16t + n_w;
  _Float16* wv16t = wk16t + n_w;
  _Float16* wo16t = wv16t + n_w;
  _Float16* qbuf  = wo16t + n_w;                      // q: [B,H,S,64]
  _Float16* kbuf  = qbuf + n_e;                       // k: [B,H,S,64]
  _Float16* vbuf  = kbuf + n_e;                       // v: [B,H,64,S] (transposed)
  _Float16* ctx   = vbuf + n_e;                       // [B,S,H*64]
  float*    xbuf  = (float*)xq;                       // alias: xq+xk reused
                                                      // after kernel 1 is done
  const int n4e = (int)(n_e / 4);
  cvt16_kernel<<<dim3((n4e + 255) / 256), 256, 0, stream>>>(Q, xq, n4e);
  cvt16_kernel<<<dim3((n4e + 255) / 256), 256, 0, stream>>>(K, xk, n4e);
  cvt16_kernel<<<dim3((n4e + 255) / 256), 256, 0, stream>>>(V, xv, n4e);
  cvtT_kernel<<<dim3(D_ / 32, D_ / 32), 256, 0, stream>>>(Wq, wq16t);
  cvtT_kernel<<<dim3(D_ / 32, D_ / 32), 256, 0, stream>>>(Wk, wk16t);
  cvtT_kernel<<<dim3(D_ / 32, D_ / 32), 256, 0, stream>>>(Wv, wv16t);
  cvtT_kernel<<<dim3(D_ / 32, D_ / 32), 256, 0, stream>>>(Wo, wo16t);

  qkv_proj_kernel<<<dim3(D_ / 128, (B_ * S_) / 128, 3), 256, 0, stream>>>(
      xq, xk, xv, wq16t, bq, wk16t, bk, wv16t, bv, qbuf, kbuf, vbuf);
  scores_kernel<<<dim3(S_ / 128, S_ / 128, B_ * H_), 256, 0, stream>>>(
      qbuf, kbuf, mask, attn);
  softmax_ctx_kernel<<<dim3(S_ / 128, B_ * H_), 256, 0, stream>>>(
      vbuf, attn, ctx);
  out_proj_kernel<<<dim3(D_ / 128, (B_ * S_) / 128), 256, 0, stream>>>(
      ctx, wo16t, bo, Q, xbuf);
  layernorm_kernel<<<dim3((B_ * S_) / 8), 256, 0, stream>>>(
      xbuf, gamma, beta, outp);
}
